// AFTGeneral_74208444940326
// MI455X (gfx1250) — matrix operand order repaired and checked by reading, not compiled, not run on hardware
//
#include <hip/hip_runtime.h>

typedef __attribute__((ext_vector_type(16))) _Float16     v16h;
typedef __attribute__((ext_vector_type(8)))  _Float16     v8h;
typedef __attribute__((ext_vector_type(8)))  float        v8f;
typedef __attribute__((ext_vector_type(4)))  unsigned int v4u;
typedef __attribute__((ext_vector_type(8)))  int          v8i;
typedef __attribute__((ext_vector_type(4)))  int          v4i;

#define NTOK 1024     // sequence length
#define DIMX 128      // hidden dim
#define LDSTRIDE 1032 // padded LDS row stride in halves (2064 B = 4 banks mod 64)

// ---------------------------------------------------------------------------
// WMMA helpers (CDNA5 gfx1250, wave32, V_WMMA_F32_16X16X32_F16)
// ---------------------------------------------------------------------------

// A-fragment: 16x32 f16 tile of row-major A[lda] at (row_base, k_base).
// ISA layout: lane<16 holds M=lane, elems 0..7 = K h*8..h*8+7, elems 8..15 =
// K 16+h*8..16+h*8+7 where h = lane/16.
__device__ __forceinline__ v16h load_a16(const _Float16* A, int lda,
                                         int row_base, int k_base, int lane) {
    int r = lane & 15, h = lane >> 4;
    const _Float16* p = A + (size_t)(row_base + r) * lda + k_base + h * 8;
    v8h lo = *(const v8h*)(p);        // K = k_base + h*8 .. +7
    v8h hi = *(const v8h*)(p + 16);   // K = k_base + 16 + h*8 .. +7
    return __builtin_shufflevector(lo, hi, 0, 1, 2, 3, 4, 5, 6, 7,
                                           8, 9, 10, 11, 12, 13, 14, 15);
}

// Same A-fragment, but from the padded LDS slab (row 0..15, stride LDSTRIDE).
__device__ __forceinline__ v16h load_a16_lds(const _Float16* A,
                                             int k_base, int lane) {
    int r = lane & 15, h = lane >> 4;
    const _Float16* p = A + r * LDSTRIDE + k_base + h * 8;
    v8h lo = *(const v8h*)(p);
    v8h hi = *(const v8h*)(p + 16);
    return __builtin_shufflevector(lo, hi, 0, 1, 2, 3, 4, 5, 6, 7,
                                           8, 9, 10, 11, 12, 13, 14, 15);
}

// B-fragment for X*Y^T: B = Y^T, Y row-major (N x K). Lane holds N=lane%16,
// elems 0..7 = K h*8..h*8+7, elems 8..15 = K 16+h*8..16+h*8+7 (h = lane/16).
__device__ __forceinline__ v16h load_bt16(const _Float16* Y, int ldy,
                                          int col_base, int k_base, int lane) {
    int r = lane & 15, h = lane >> 4;
    const _Float16* p = Y + (size_t)(col_base + r) * ldy + k_base + h * 8;
    v8h lo = *(const v8h*)(p);
    v8h hi = *(const v8h*)(p + 16);
    return __builtin_shufflevector(lo, hi, 0, 1, 2, 3, 4, 5, 6, 7,
                                           8, 9, 10, 11, 12, 13, 14, 15);
}

__device__ __forceinline__ v8f wmma16(v16h a, v16h b, v8f c) {
    return __builtin_amdgcn_wmma_f32_16x16x32_f16(
        /*neg_a=*/false, a, /*neg_b=*/false, b,
        /*c_mod=*/(short)0, c, /*reuse_a=*/false, /*reuse_b=*/false);
}

__device__ __forceinline__ float sigmoidf_(float x) {
    return 1.0f / (1.0f + __expf(-x));
}

// ---------------------------------------------------------------------------
// Stage A: fp32 -> fp16 conversion
// ---------------------------------------------------------------------------
__global__ void cvt_f32_f16(const float* __restrict__ in,
                            _Float16* __restrict__ out, int n) {
    int i = blockIdx.x * blockDim.x + threadIdx.x;
    if (i < n) out[i] = (_Float16)in[i];
}

// ---------------------------------------------------------------------------
// Stage B: q/k/v projections + epilogue.
//   q  = sigmoid(x Wq^T)                    (f32 [N][D])
//   T[d][m]     = exp(k[m][d])*val[m][d]    (f16 transposed, ld = NTOK)
//   T[128+d][m] = exp(k[m][d])
// grid: 64 blocks (m tiles) x 256 threads (8 waves = 8 d tiles)
// ---------------------------------------------------------------------------
__global__ __launch_bounds__(256) void proj_kernel(
    const _Float16* __restrict__ xh,  const _Float16* __restrict__ Wqh,
    const _Float16* __restrict__ Wkh, const _Float16* __restrict__ Wvh,
    float* __restrict__ qf, _Float16* __restrict__ T) {
    int lane = threadIdx.x & 31;
    int wave = threadIdx.x >> 5;
    int mb = blockIdx.x * 16;   // token tile
    int db = wave * 16;         // channel tile

    v8f aq = {}, ak = {}, av = {};
    for (int kb = 0; kb < DIMX; kb += 32) {
        v16h a  = load_a16 (xh,  DIMX, mb, kb, lane);
        v16h bq = load_bt16(Wqh, DIMX, db, kb, lane);
        v16h bk = load_bt16(Wkh, DIMX, db, kb, lane);
        v16h bv = load_bt16(Wvh, DIMX, db, kb, lane);
        aq = wmma16(a, bq, aq);
        ak = wmma16(a, bk, ak);
        av = wmma16(a, bv, av);
    }

    int r = lane & 15, h = lane >> 4;
    int d = db + r;
    #pragma unroll
    for (int g = 0; g < 8; ++g) {
        int m = mb + g + 8 * h;
        qf[(size_t)m * DIMX + d] = sigmoidf_(aq[g]);
    }
    v8h ekv_pack, ek_pack;
    #pragma unroll
    for (int g = 0; g < 8; ++g) {
        float e = __expf(ak[g]);
        ek_pack[g]  = (_Float16)e;
        ekv_pack[g] = (_Float16)(e * av[g]);
    }
    int mcol = mb + 8 * h;
    *(v8h*)(T + (size_t)d * NTOK + mcol)          = ekv_pack;
    *(v8h*)(T + (size_t)(DIMX + d) * NTOK + mcol) = ek_pack;
}

// ---------------------------------------------------------------------------
// Stage C: Wmat[n][m] = exp( (u v^T)[n][m] )   (f16 [N][N])
// grid: 512 blocks x 8 waves = 4096 tiles
// ---------------------------------------------------------------------------
__global__ __launch_bounds__(256) void posbias_kernel(
    const _Float16* __restrict__ uh, const _Float16* __restrict__ vh,
    _Float16* __restrict__ Wmat) {
    int lane = threadIdx.x & 31;
    int wave = threadIdx.x >> 5;
    int tile = blockIdx.x * 8 + wave;
    int nb = (tile >> 6) * 16;
    int mb = (tile & 63) * 16;

    v8f acc = {};
    for (int kb = 0; kb < DIMX; kb += 32) {
        v16h a = load_a16 (uh, DIMX, nb, kb, lane);
        v16h b = load_bt16(vh, DIMX, mb, kb, lane);
        acc = wmma16(a, b, acc);
    }

    int r = lane & 15, h = lane >> 4;
    #pragma unroll
    for (int g = 0; g < 8; ++g) {
        int n = nb + g + 8 * h;
        Wmat[(size_t)n * NTOK + mb + r] = (_Float16)__expf(acc[g]);
    }
}

// ---------------------------------------------------------------------------
// Stage D: nd[n][c] = (Wmat * T^T)[n][c]  (f32 [N][256]; c<128 num, c>=128 den)
// One 16-row A-slab of Wmat (16 x 1024 f16 = 32 KB) is shared by all 8 waves
// of a block -> DMA it to LDS once via the Tensor Data Mover (iterate mode
// inserts an 8-half pad per row so ds_load_b128 A-fragment reads are
// bank-conflict-free), then the K-loop reads A from LDS and streams B from L2.
// grid: 128 blocks x 8 waves = 1024 tiles; K = 1024 (32 WMMA steps)
// ---------------------------------------------------------------------------
__global__ __launch_bounds__(256) void numden_kernel(
    const _Float16* __restrict__ Wmat, const _Float16* __restrict__ T,
    float* __restrict__ nd) {
    __shared__ _Float16 lA[16 * LDSTRIDE];   // 33 KB

    int lane = threadIdx.x & 31;
    int wave = threadIdx.x >> 5;
    int nb = (blockIdx.x >> 1) * 16;                  // shared by all 8 waves
    int cb = (((blockIdx.x & 1) << 3) + wave) * 16;   // per-wave col tile

    if (wave == 0) {
        // ---- Tensor DMA: Wmat[nb..nb+15][0..1023] -> lA (padded rows) ----
        unsigned long long ga =
            (unsigned long long)(size_t)(Wmat + (size_t)nb * NTOK);
        unsigned int ldsOff = (unsigned int)(size_t)(&lA[0]); // low32 = LDS offset

        // D# group0: count=1 | lds_addr | global_addr[56:0] | type=2
        v4u g0 = { 1u,
                   ldsOff,
                   (unsigned int)(ga & 0xFFFFFFFFu),
                   (unsigned int)((ga >> 32) & 0x01FFFFFFu) | 0x80000000u };
        // D# group1: data_size=2B (1<<16), iterate_enable (1<<19);
        // tensor_dim0=1024, tensor_dim1=1, tile_dim0=1024, tile_dim1=1,
        // tensor_dim0_stride=1024, tensor_dim1_stride=1024
        v8i g1 = { (int)((1u << 16) | (1u << 19)),
                   (int)(1024u << 16),   // tensor_dim0[15:0] @ bits 79:48
                   (int)(1u    << 16),   // tensor_dim1[15:0] @ bits 111:80
                   (int)(1024u << 16),   // tile_dim0         @ bits 127:112
                   1,                    // tile_dim1
                   1024,                 // tensor_dim0_stride[31:0]
                   (int)(1024u << 16),   // tensor_dim1_stride[15:0]
                   0 };
        // D# group2 (iterate mode): lds_addr_increment=LDSTRIDE elems,
        // global_addr_increment=1024 elems, iterate_count=15 (-> 16 rows)
        v4i g2 = { 0, LDSTRIDE, 1024, (int)(15u << 16) };
        v4i g3 = { 0, 0, 0, 0 };
        v8i g4 = { 0, 0, 0, 0, 0, 0, 0, 0 };   // extra group (clang-23 6-arg form)

        __builtin_amdgcn_tensor_load_to_lds(g0, g1, g2, g3, g4, /*cpol=*/0);
        __builtin_amdgcn_s_wait_tensorcnt(0);
    }
    __syncthreads();

    v8f acc = {};
    for (int kb = 0; kb < NTOK; kb += 32) {
        v16h a = load_a16_lds(lA, kb, lane);          // ds_load_b128 x2
        v16h b = load_bt16(T, NTOK, cb, kb, lane);    // global_load_b128 x2
        acc = wmma16(a, b, acc);
    }

    int r = lane & 15, h = lane >> 4;
    #pragma unroll
    for (int g = 0; g < 8; ++g) {
        int n = nb + g + 8 * h;
        nd[(size_t)n * 256 + cb + r] = acc[g];
    }
}

// ---------------------------------------------------------------------------
// Stage E1: s[n][d] = f16( q[n][d] * num[n][d] / den[n][d] )
// ---------------------------------------------------------------------------
__global__ void ctx_kernel(const float* __restrict__ qf,
                           const float* __restrict__ nd,
                           _Float16* __restrict__ sh) {
    int i = blockIdx.x * blockDim.x + threadIdx.x;
    if (i >= NTOK * DIMX) return;
    int n = i >> 7, d = i & (DIMX - 1);
    float num = nd[(size_t)n * 256 + d];
    float den = nd[(size_t)n * 256 + DIMX + d];
    sh[i] = (_Float16)(qf[i] * num / den);
}

// ---------------------------------------------------------------------------
// Stage E2: out = s Wo^T + bo   (f32 output)
// ---------------------------------------------------------------------------
__global__ __launch_bounds__(256) void out_kernel(
    const _Float16* __restrict__ sh, const _Float16* __restrict__ Woh,
    const float* __restrict__ bo, float* __restrict__ out) {
    int lane = threadIdx.x & 31;
    int wave = threadIdx.x >> 5;
    int mb = blockIdx.x * 16;
    int db = wave * 16;

    v8f acc = {};
    for (int kb = 0; kb < DIMX; kb += 32) {
        v16h a = load_a16 (sh,  DIMX, mb, kb, lane);
        v16h b = load_bt16(Woh, DIMX, db, kb, lane);
        acc = wmma16(a, b, acc);
    }

    int r = lane & 15, h = lane >> 4;
    int d = db + r;
    float bias = bo[d];
    #pragma unroll
    for (int g = 0; g < 8; ++g) {
        int m = mb + g + 8 * h;
        out[(size_t)m * DIMX + d] = acc[g] + bias;
    }
}

// ---------------------------------------------------------------------------
// Host launch
// ---------------------------------------------------------------------------
extern "C" void kernel_launch(void* const* d_in, const int* in_sizes, int n_in,
                              void* d_out, int out_size, void* d_ws, size_t ws_size,
                              hipStream_t stream) {
    const float* x  = (const float*)d_in[0];   // [1,1024,128]
    const float* Wq = (const float*)d_in[1];   // [128,128]
    const float* Wk = (const float*)d_in[2];
    const float* Wv = (const float*)d_in[3];
    const float* Wo = (const float*)d_in[4];
    const float* bo = (const float*)d_in[5];   // [128]
    const float* u  = (const float*)d_in[6];   // [1024,128]
    const float* v  = (const float*)d_in[7];   // [1024,128]
    float* out = (float*)d_out;                // [1,1024,128] f32

    char* ws = (char*)d_ws;
    _Float16* xh   = (_Float16*)(ws + 0);         // 256 KB
    _Float16* Wqh  = (_Float16*)(ws + 262144);    //  32 KB
    _Float16* Wkh  = (_Float16*)(ws + 294912);
    _Float16* Wvh  = (_Float16*)(ws + 327680);
    _Float16* Woh  = (_Float16*)(ws + 360448);
    _Float16* uh   = (_Float16*)(ws + 393216);    // 256 KB
    _Float16* vh   = (_Float16*)(ws + 655360);    // 256 KB
    float*    qf   = (float*)   (ws + 917504);    // 512 KB
    _Float16* T    = (_Float16*)(ws + 1441792);   // 512 KB  [256][1024] f16
    _Float16* Wm   = (_Float16*)(ws + 1966080);   //   2 MB  [1024][1024] f16
    float*    nd   = (float*)   (ws + 4063232);   //   1 MB  [1024][256] f32
    _Float16* sh   = (_Float16*)(ws + 5111808);   // 256 KB
    (void)ws_size; (void)n_in; (void)in_sizes; (void)out_size;

    cvt_f32_f16<<<512, 256, 0, stream>>>(x,  xh,  NTOK * DIMX);
    cvt_f32_f16<<< 64, 256, 0, stream>>>(Wq, Wqh, DIMX * DIMX);
    cvt_f32_f16<<< 64, 256, 0, stream>>>(Wk, Wkh, DIMX * DIMX);
    cvt_f32_f16<<< 64, 256, 0, stream>>>(Wv, Wvh, DIMX * DIMX);
    cvt_f32_f16<<< 64, 256, 0, stream>>>(Wo, Woh, DIMX * DIMX);
    cvt_f32_f16<<<512, 256, 0, stream>>>(u,  uh,  NTOK * DIMX);
    cvt_f32_f16<<<512, 256, 0, stream>>>(v,  vh,  NTOK * DIMX);

    proj_kernel   <<< 64, 256, 0, stream>>>(xh, Wqh, Wkh, Wvh, qf, T);
    posbias_kernel<<<512, 256, 0, stream>>>(uh, vh, Wm);
    numden_kernel <<<128, 256, 0, stream>>>(Wm, T, nd);
    ctx_kernel    <<<512, 256, 0, stream>>>(qf, nd, sh);
    out_kernel    <<< 64, 256, 0, stream>>>(sh, Woh, bo, out);
}
